// HGNet_78091095375881
// MI455X (gfx1250) — matrix operand order repaired and checked by reading, not compiled, not run on hardware
//
#include <hip/hip_runtime.h>
#include <hip/hip_bf16.h>

// ---------- types for WMMA ----------
typedef __bf16 bf16_t;
typedef bf16_t v16bf __attribute__((ext_vector_type(16)));
typedef float  v8f   __attribute__((ext_vector_type(8)));

union FragAB { v16bf v; uint4 q[2]; };

__device__ __forceinline__ unsigned short f2bf(float f) {
    union { float f; unsigned int u; } x; x.f = f;
    unsigned int u = x.u + 0x7FFFu + ((x.u >> 16) & 1u);   // round-to-nearest-even
    return (unsigned short)(u >> 16);
}
__device__ __forceinline__ float bf2f(unsigned short h) {
    union { unsigned int u; float f; } x; x.u = ((unsigned int)h) << 16;
    return x.f;
}
__device__ __forceinline__ unsigned int pack2(float a, float b) {
    return (unsigned int)f2bf(a) | ((unsigned int)f2bf(b) << 16);
}
__device__ __forceinline__ float sigmoidf(float t) { return 1.0f / (1.0f + __expf(-t)); }
__device__ __forceinline__ float lrelu(float z)    { return z > 0.f ? z : 0.01f * z; }
__device__ __forceinline__ float relu(float z)     { return z > 0.f ? z : 0.f; }

// ---------- CDNA5 async global->LDS (ASYNCcnt-tracked DMA), guarded ----------
#if __has_builtin(__builtin_amdgcn_global_load_async_to_lds_b128)
#define HAS_ASYNC_LDS 1
#else
#define HAS_ASYNC_LDS 0
#endif

#if HAS_ASYNC_LDS
typedef int vint4_t __attribute__((vector_size(16)));
typedef __attribute__((address_space(1))) vint4_t* gv4_ptr;
typedef __attribute__((address_space(3))) vint4_t* lv4_ptr;
#endif

__device__ __forceinline__ void async_copy16(const void* g, void* l) {
#if HAS_ASYNC_LDS
    __builtin_amdgcn_global_load_async_to_lds_b128((gv4_ptr)g, (lv4_ptr)l, 0, 0);
#else
    *(uint4*)l = *(const uint4*)g;   // fallback: VGPR round trip
#endif
}
__device__ __forceinline__ void async_wait() {
#if HAS_ASYNC_LDS
#if __has_builtin(__builtin_amdgcn_s_wait_asynccnt)
    __builtin_amdgcn_s_wait_asynccnt(0);
#else
    asm volatile("s_wait_asynccnt 0x0" ::: "memory");
#endif
#endif
}

// =====================================================================
// 1) Differentiable histogram: per (b,c) plane, 20 soft bins.
//    Deterministic: per-block partials, summed later in fixed order.
// =====================================================================
__global__ __launch_bounds__(256)
void hist_kernel(const float* __restrict__ x, float* __restrict__ part) {
    const int tid = threadIdx.x;
    const int blk = blockIdx.x, c = blockIdx.y, b = blockIdx.z;
    const float* plane = x + ((size_t)(b * 3 + c)) * (512 * 512);
    float acc[20];
#pragma unroll
    for (int k = 0; k < 20; ++k) acc[k] = 0.f;
    for (int idx = blk * 256 + tid; idx < 512 * 512; idx += 64 * 256) {
        float xv = plane[idx];
        float prev = sigmoidf(100.0f * xv);
#pragma unroll
        for (int i = 1; i <= 20; ++i) {
            float cur = sigmoidf(100.0f * (xv - 0.05f * (float)i));
            acc[i - 1] += prev - cur;
            prev = cur;
        }
    }
    __shared__ float wred[8][20];
    const int lane = tid & 31, wv = tid >> 5;
#pragma unroll
    for (int k = 0; k < 20; ++k) {
        float v = acc[k];
        for (int off = 16; off > 0; off >>= 1) v += __shfl_down(v, off, 32);
        if (lane == 0) wred[wv][k] = v;
    }
    __syncthreads();
    if (tid < 20) {
        float s = 0.f;
#pragma unroll
        for (int w = 0; w < 8; ++w) s += wred[w][tid];
        part[(((b * 3 + c) * 64) + blk) * 20 + tid] = s;
    }
}

// =====================================================================
// 2) Tiny MLP 60->128->256->16->5168, one block, LDS-resident.
// =====================================================================
__global__ __launch_bounds__(256)
void mlp_kernel(const float* __restrict__ part,
                const float* __restrict__ W1, const float* __restrict__ b1,
                const float* __restrict__ W2, const float* __restrict__ b2,
                const float* __restrict__ W3, const float* __restrict__ b3,
                const float* __restrict__ W4, const float* __restrict__ b4,
                float* __restrict__ p) {
    __shared__ float h0[4 * 60];
    __shared__ float a1[4 * 128];
    __shared__ float a2[4 * 256];
    __shared__ float a3[4 * 16];
    const int tid = threadIdx.x;
    for (int o = tid; o < 240; o += 256) {
        int b = o / 60, r = o % 60, bin = r / 3, c = r % 3;
        float s = 0.f;
        const float* src = part + ((b * 3 + c) * 64) * 20 + bin;
        for (int j = 0; j < 64; ++j) s += src[j * 20];
        h0[o] = s * (1.0f / 262144.0f);
    }
    __syncthreads();
    for (int o = tid; o < 4 * 128; o += 256) {
        int b = o >> 7, j = o & 127;
        float acc = b1[j];
        for (int i = 0; i < 60; ++i) acc += h0[b * 60 + i] * W1[i * 128 + j];
        a1[o] = lrelu(acc);
    }
    __syncthreads();
    for (int o = tid; o < 4 * 256; o += 256) {
        int b = o >> 8, j = o & 255;
        float acc = b2[j];
        for (int i = 0; i < 128; ++i) acc += a1[b * 128 + i] * W2[i * 256 + j];
        a2[o] = lrelu(acc);
    }
    __syncthreads();
    for (int o = tid; o < 4 * 16; o += 256) {
        int b = o >> 4, j = o & 15;
        float acc = b3[j];
        for (int i = 0; i < 256; ++i) acc += a2[b * 256 + i] * W3[i * 16 + j];
        a3[o] = lrelu(acc);
    }
    __syncthreads();
    for (int o = tid; o < 4 * 5168; o += 256) {
        int b = o / 5168, j = o % 5168;
        float acc = b4[j];
#pragma unroll
        for (int i = 0; i < 16; ++i) acc += a3[b * 16 + i] * W4[i * 5168 + j];
        p[o] = acc;
    }
}

// =====================================================================
// 3) conv0: 3->32, 3x3, SAME, relu. K=27 too small for WMMA; direct.
//    Output bf16 NHWC (b,y,x,ci) -> 4x b128 stores per pixel.
// =====================================================================
__global__ __launch_bounds__(256)
void conv0_kernel(const float* __restrict__ x, unsigned short* __restrict__ out,
                  const float* __restrict__ W, const float* __restrict__ bias) {
    __shared__ float sw[32 * 27];
    __shared__ float sb[32];
    const int tid = threadIdx.x;
    for (int i = tid; i < 32 * 27; i += 256) sw[i] = W[i];
    if (tid < 32) sb[tid] = bias[tid];
    __syncthreads();
    const int px = blockIdx.x * 256 + tid, y = blockIdx.y, b = blockIdx.z;
    float v[27];
#pragma unroll
    for (int ci = 0; ci < 3; ++ci)
#pragma unroll
        for (int dy = 0; dy < 3; ++dy)
#pragma unroll
            for (int dx = 0; dx < 3; ++dx) {
                int iy = y + dy - 1, ix = px + dx - 1;
                float t = 0.f;
                if ((unsigned)iy < 512u && (unsigned)ix < 512u)
                    t = x[(((size_t)(b * 3 + ci)) * 512 + iy) * 512 + ix];
                v[(ci * 3 + dy) * 3 + dx] = t;
            }
    float o[32];
#pragma unroll
    for (int co = 0; co < 32; ++co) {
        float acc = sb[co];
#pragma unroll
        for (int k = 0; k < 27; ++k) acc += v[k] * sw[co * 27 + k];
        o[co] = relu(acc);
    }
    unsigned short* op = &out[(((size_t)b * 512 + y) * 512 + px) * 32];
#pragma unroll
    for (int g = 0; g < 4; ++g) {
        uint4 st;
        st.x = pack2(o[g * 8 + 0], o[g * 8 + 1]);
        st.y = pack2(o[g * 8 + 2], o[g * 8 + 3]);
        st.z = pack2(o[g * 8 + 4], o[g * 8 + 5]);
        st.w = pack2(o[g * 8 + 6], o[g * 8 + 7]);
        *(uint4*)(op + g * 8) = st;
    }
}

// =====================================================================
// 4) Implicit-GEMM conv on v_wmma_f32_16x16x32_bf16, NHWC activations.
//    K ordered (tap, ci): each 3x3 tap = one K=32 WMMA step.
//    Block tile: COUT x 64 px x 4 rows (weights staged once per block).
//    Patch staged with GLOBAL_LOAD_ASYNC_TO_LDS_B128 (16B chunks land
//    directly in the swizzled fragment layout; NHWC makes them
//    globally contiguous). Fragment layouts per ISA:
//      A 16x32 bf16: lanes0-15 v0-3=K0..7, v4-7=K16..23;
//                    lanes16-31 v0-3=K8..15, v4-7=K24..31
//      B 32x16:      same K order as A: lanes0-15 K0..7,K16..23; lanes16-31 K8..15,K24..31
//      C/D f32:      lane=N(pixel), vgpr r=M(co), +8 for hi lanes
// =====================================================================
template <int CIN, int COUT, int TAPS>
__global__ __launch_bounds__(32 * (COUT / 16) * 4)
void conv_wmma(const unsigned short* __restrict__ in, unsigned short* __restrict__ out,
               const float* __restrict__ w, const float* __restrict__ bias,
               int bstride) {
    constexpr int H = 512, W = 512;
    constexpr int YB = 4;                       // output rows per block
    constexpr int HALO = (TAPS == 9) ? 1 : 0;
    constexpr int ROWS_IN = YB + 2 * HALO;
    constexpr int XT = 64 + 2 * HALO;
    constexpr int NW = (COUT / 16) * 4;         // waves per block
    constexpr int NT = NW * 32;
    constexpr int CCH = CIN / 8;                // valid 8-ci chunks per pixel

    __shared__ __align__(16) unsigned short sA[TAPS * COUT * 32]; // [tap][co][chunk^sw][8]
    __shared__ __align__(16) unsigned short sP[ROWS_IN * XT * 32];// [dy][xx][chunk^sw][8]
    __shared__ float sB[COUT];

    const int tid = threadIdx.x;
    const int b = blockIdx.z, y0 = blockIdx.y * YB, x0 = blockIdx.x * 64;
    const float* wb = w + b * bstride;
    const float* bbias = bias + b * bstride;
    if (tid < COUT) sB[tid] = bbias[tid];

    // weights -> LDS bf16, zero-padded ci>=CIN, A-fragment swizzle on co
    for (int e = tid; e < TAPS * COUT * 32; e += NT) {
        int ci = e & 31; int t2 = e >> 5; int co = t2 % COUT; int tap = t2 / COUT;
        float v = (ci < CIN) ? wb[(co * CIN + ci) * TAPS + tap] : 0.f;
        sA[((tap * COUT + co) * 4 + ((ci >> 3) ^ (co & 3))) * 8 + (ci & 7)] = f2bf(v);
    }

    // input patch: 16B chunks; async DMA for in-bounds, ds-store zeros for pad
    for (int q = tid; q < ROWS_IN * XT * 4; q += NT) {
        int c8 = q & 3; int t = q >> 2; int xx = t % XT; int dy = t / XT;
        int iy = y0 + dy - HALO;
        int ix = x0 + xx - HALO;
        unsigned short* lp = &sP[((dy * XT + xx) * 4 + (c8 ^ (xx & 3))) * 8];
        if (c8 < CCH && (unsigned)iy < (unsigned)H && (unsigned)ix < (unsigned)W) {
            const unsigned short* gp = &in[(((size_t)b * H + iy) * W + ix) * CIN + c8 * 8];
            async_copy16(gp, lp);
        } else {
            uint4 z = {0, 0, 0, 0};
            *(uint4*)lp = z;
        }
    }
    async_wait();
    __syncthreads();

    const int wv = tid >> 5, lane = tid & 31;
    const int co_blk = wv >> 2, n_blk = wv & 3;
    const int m = lane & 15, hi = lane >> 4;
    const int xo = x0 + n_blk * 16 + m;
    const int co0 = co_blk * 16 + hi * 8;       // 8 consecutive co held by this lane

#pragma unroll
    for (int yo = 0; yo < YB; ++yo) {
        v8f acc = {};
#pragma unroll
        for (int tap = 0; tap < TAPS; ++tap) {
            const int ky = (TAPS == 9) ? (tap / 3) : 0;
            const int kx = (TAPS == 9) ? (tap % 3) : 0;
            FragAB A, Bf;
            {
                int co = co_blk * 16 + m;
                int abase = (tap * COUT + co) * 4;
                int c0 = hi ? 1 : 0, c1 = hi ? 3 : 2;
                A.q[0] = *(const uint4*)&sA[(abase + (c0 ^ (co & 3))) * 8];
                A.q[1] = *(const uint4*)&sA[(abase + (c1 ^ (co & 3))) * 8];
            }
            {
                int xx = n_blk * 16 + m + kx;   // patch x (HALO folded into staging)
                int dy = yo + ky;
                int bbase = (dy * XT + xx) * 4;
                int c0 = hi ? 1 : 0, c1 = hi ? 3 : 2;
                Bf.q[0] = *(const uint4*)&sP[(bbase + (c0 ^ (xx & 3))) * 8];
                Bf.q[1] = *(const uint4*)&sP[(bbase + (c1 ^ (xx & 3))) * 8];
            }
            acc = __builtin_amdgcn_wmma_f32_16x16x32_bf16(
                false, A.v, false, Bf.v, (short)0, acc, false, false);
        }
        // bias + relu, pack 8 consecutive co, one b128 store (NHWC)
        uint4 st;
        st.x = pack2(relu(acc[0] + sB[co0 + 0]), relu(acc[1] + sB[co0 + 1]));
        st.y = pack2(relu(acc[2] + sB[co0 + 2]), relu(acc[3] + sB[co0 + 3]));
        st.z = pack2(relu(acc[4] + sB[co0 + 4]), relu(acc[5] + sB[co0 + 5]));
        st.w = pack2(relu(acc[6] + sB[co0 + 6]), relu(acc[7] + sB[co0 + 7]));
        unsigned short* op = &out[(((size_t)b * H + (y0 + yo)) * W + xo) * COUT + co0];
        *(uint4*)op = st;
    }
}

// =====================================================================
// 5) Wout conv (16->3, 3x3, no relu, NHWC input) fused with _curve.
//    d_out[0 .. 3.1M) = curve, d_out[3.1M .. 6.3M) = conv out (NCHW f32).
// =====================================================================
__global__ __launch_bounds__(256)
void final_kernel(const unsigned short* __restrict__ act,
                  const float* __restrict__ x,
                  const float* __restrict__ W, const float* __restrict__ bias,
                  float* __restrict__ out) {
    __shared__ float sw[3 * 16 * 9];
    __shared__ float sb[3];
    const int tid = threadIdx.x;
    for (int i = tid; i < 432; i += 256) sw[i] = W[i];
    if (tid < 3) sb[tid] = bias[tid];
    __syncthreads();
    const int px = blockIdx.x * 256 + tid, y = blockIdx.y, b = blockIdx.z;
    float a0 = sb[0], a1 = sb[1], a2 = sb[2];
#pragma unroll
    for (int dy = 0; dy < 3; ++dy) {
#pragma unroll
        for (int dx = 0; dx < 3; ++dx) {
            const int t = dy * 3 + dx;
            const int iy = y + dy - 1, ix = px + dx - 1;
            if ((unsigned)iy < 512u && (unsigned)ix < 512u) {
                const uint4* ap = (const uint4*)&act[(((size_t)b * 512 + iy) * 512 + ix) * 16];
                uint4 q0 = ap[0], q1 = ap[1];
                unsigned int uu[8] = {q0.x, q0.y, q0.z, q0.w, q1.x, q1.y, q1.z, q1.w};
#pragma unroll
                for (int ci = 0; ci < 16; ++ci) {
                    unsigned int wrd = uu[ci >> 1];
                    float v = bf2f((unsigned short)((ci & 1) ? (wrd >> 16) : (wrd & 0xffffu)));
                    a0 += v * sw[(0 * 16 + ci) * 9 + t];
                    a1 += v * sw[(1 * 16 + ci) * 9 + t];
                    a2 += v * sw[(2 * 16 + ci) * 9 + t];
                }
            }
        }
    }
    // _curve: s=5 => A = max_y - min_y, Bc = min_y
    const float A  = 0.98661429f;
    const float Bc = 0.00669285f;
    const size_t OUT2 = (size_t)4 * 3 * 512 * 512;
    float al[3] = {a0, a1, a2};
#pragma unroll
    for (int c = 0; c < 3; ++c) {
        size_t idx = (((size_t)(b * 3 + c)) * 512 + y) * 512 + px;
        out[OUT2 + idx] = al[c];
        float xv = x[idx];
        float yv = 1.0f / (A * xv + Bc) - 1.0f;
        yv = 1.0f + yv * __expf(-10.0f * al[c]);
        out[idx] = 1.0f / (A * yv) - Bc / A;
    }
}

// =====================================================================
// Launch
// =====================================================================
extern "C" void kernel_launch(void* const* d_in, const int* in_sizes, int n_in,
                              void* d_out, int out_size, void* d_ws, size_t ws_size,
                              hipStream_t stream) {
    (void)in_sizes; (void)n_in; (void)out_size; (void)ws_size;
    const float* x_img = (const float*)d_in[0];
    const float* W1 = (const float*)d_in[1];  const float* b1 = (const float*)d_in[2];
    const float* W2 = (const float*)d_in[3];  const float* b2 = (const float*)d_in[4];
    const float* W3 = (const float*)d_in[5];  const float* b3 = (const float*)d_in[6];
    const float* W4 = (const float*)d_in[7];  const float* b4 = (const float*)d_in[8];
    const float* Wc0 = (const float*)d_in[9]; const float* bc0 = (const float*)d_in[10];
    const float* Wc1 = (const float*)d_in[11];const float* bc1 = (const float*)d_in[12];
    const float* Wout = (const float*)d_in[13];const float* bout = (const float*)d_in[14];
    float* out = (float*)d_out;

    char* ws = (char*)d_ws;
    float* part = (float*)(ws);                        // 4*3*64*20 f32 = 60 KiB
    float* p    = (float*)(ws + (64 * 1024));          // 4*5168 f32   ~ 81 KiB
    const size_t ACT_BYTES = (size_t)4 * 32 * 512 * 512 * 2;  // 64 MiB bf16
    unsigned short* actA = (unsigned short*)(ws + (256 * 1024));
    unsigned short* actB = (unsigned short*)(ws + (256 * 1024) + ACT_BYTES);

    hist_kernel<<<dim3(64, 3, 4), 256, 0, stream>>>(x_img, part);
    mlp_kernel<<<1, 256, 0, stream>>>(part, W1, b1, W2, b2, W3, b3, W4, b4, p);
    conv0_kernel<<<dim3(2, 512, 4), 256, 0, stream>>>(x_img, actA, Wc0, bc0);
    // static 32->32 3x3
    conv_wmma<32, 32, 9><<<dim3(8, 128, 4), 256, 0, stream>>>(actA, actB, Wc1, bc1, 0);
    // dynamic 32->16 3x3 : w2 = p[b][0:4608], bb2 = p[b][4608:4624]
    conv_wmma<32, 16, 9><<<dim3(8, 128, 4), 128, 0, stream>>>(actB, actA, p + 0, p + 4608, 5168);
    // dynamic 16->16 1x1 : w3 = p[b][4624:4880], bb3 = p[b][4880:4896]
    conv_wmma<16, 16, 1><<<dim3(8, 128, 4), 128, 0, stream>>>(actA, actB, p + 4624, p + 4880, 5168);
    // dynamic 16->16 1x1 : w4 = p[b][4896:5152], bb4 = p[b][5152:5168]
    conv_wmma<16, 16, 1><<<dim3(8, 128, 4), 128, 0, stream>>>(actB, actA, p + 4896, p + 5152, 5168);
    final_kernel<<<dim3(2, 512, 4), 256, 0, stream>>>(actA, x_img, Wout, bout, out);
}
